// Transformer_56676388438131
// MI455X (gfx1250) — matrix operand order repaired and checked by reading, not compiled, not run on hardware
//
#include <hip/hip_runtime.h>
#include <hip/hip_bf16.h>

// Problem constants (match reference)
#define BB 8
#define CC 512
#define NN 2048
#define DD 64

typedef __attribute__((ext_vector_type(16))) __bf16 v16bf;
typedef __attribute__((ext_vector_type(8)))  float  v8f;
typedef __attribute__((ext_vector_type(4)))  float  f32x4;
typedef __attribute__((ext_vector_type(4)))  unsigned int u32x4;

union Frag16 {
    v16bf  v;
    __bf16 h[16];
    u32x4  q[2];
};

// f32 -> bf16: native conversion (v_cvt_pk_bf16_f32 on gfx1250)
static __device__ __forceinline__ __bf16 f2bf(float f) { return (__bf16)f; }

// Fragment loader: two contiguous 16B chunks (elements 0..7 from p0, 8..15 from p1)
static __device__ __forceinline__ v16bf ld_frag(const __bf16* p0, const __bf16* p1) {
    Frag16 f;
    f.q[0] = *(const u32x4*)p0;
    f.q[1] = *(const u32x4*)p1;
    return f.v;
}

#define WMMA_BF16(A, B, C_) \
    __builtin_amdgcn_wmma_f32_16x16x32_bf16(false, (A), false, (B), (short)0, (C_), false, false)

// -------------------------------------------------------------------------
// Prep 0: elementwise f32 -> bf16 (weights). Each thread converts 8.
// -------------------------------------------------------------------------
__global__ __launch_bounds__(256) void k_cvt_bf16(
        const float* __restrict__ src, __bf16* __restrict__ dst, int n) {
    const int i = (blockIdx.x * 256 + threadIdx.x) * 8;
    if (i >= n) return;
    const f32x4* s = (const f32x4*)(src + i);
    f32x4 a = s[0], b = s[1];
    Frag16 f;
    f.h[0] = f2bf(a[0]); f.h[1] = f2bf(a[1]); f.h[2] = f2bf(a[2]); f.h[3] = f2bf(a[3]);
    f.h[4] = f2bf(b[0]); f.h[5] = f2bf(b[1]); f.h[6] = f2bf(b[2]); f.h[7] = f2bf(b[3]);
    *(u32x4*)(dst + i) = f.q[0];
}

// -------------------------------------------------------------------------
// Prep 1: xT[b][n][c] (bf16) = transpose of x[b][c][n], via LDS 64x64 tile.
// grid (N/64, C/64, B), 256 threads.
// -------------------------------------------------------------------------
__global__ __launch_bounds__(256) void k_transpose_x(
        const float* __restrict__ x, __bf16* __restrict__ xT) {
    __shared__ float t[64 * 65];
    const int b  = blockIdx.z;
    const int nb = blockIdx.x * 64;
    const int cb = blockIdx.y * 64;
    const int tid = threadIdx.x;

    {   // coalesced read along n
        const int cl = tid >> 2;
        const int nl = (tid & 3) * 16;
        const f32x4* s4 = (const f32x4*)(x + ((size_t)(b * CC + cb + cl)) * NN + nb + nl);
#pragma unroll
        for (int j = 0; j < 4; ++j) {
            f32x4 v = s4[j];
            t[cl * 65 + nl + j * 4 + 0] = v[0];
            t[cl * 65 + nl + j * 4 + 1] = v[1];
            t[cl * 65 + nl + j * 4 + 2] = v[2];
            t[cl * 65 + nl + j * 4 + 3] = v[3];
        }
    }
    __syncthreads();
    {   // transposed write, contiguous 32B per thread along c
        const int nl = tid >> 2;
        const int cl = (tid & 3) * 16;
        Frag16 f;
#pragma unroll
        for (int j = 0; j < 16; ++j) f.h[j] = f2bf(t[(cl + j) * 65 + nl]);
        u32x4* dst = (u32x4*)(xT + ((size_t)b * NN + nb + nl) * CC + cb + cl);
        dst[0] = f.q[0];
        dst[1] = f.q[1];
    }
}

// -------------------------------------------------------------------------
// Kernel 1a: fT[b][n][d] = (Wq x)[d,n],  gT likewise.  Software-pipelined:
// K-loop double-buffers 1 A fragment (xT row) + 4 B fragments (Wq/Wk rows).
// grid (N/64, B), 256 threads; wave w: n-subtile (w>>1), d-half (w&1).
// -------------------------------------------------------------------------
__global__ __launch_bounds__(256) void k_proj_fg(
        const __bf16* __restrict__ xT, const __bf16* __restrict__ Wqb,
        const __bf16* __restrict__ Wkb,
        __bf16* __restrict__ fT, __bf16* __restrict__ gT) {
    const int b  = blockIdx.y;
    const int nb = blockIdx.x * 64;
    const int tid = threadIdx.x;
    const int w = tid >> 5, lane = tid & 31, lm = lane & 15, g = lane >> 4;
    const int nsub = w >> 1;            // 0..3
    const int dh   = (w & 1) * 2;       // d-tiles dh, dh+1

    v8f accF[2] = {};
    v8f accG[2] = {};

    const __bf16* xrow = xT + ((size_t)b * NN + nb + nsub * 16 + lm) * CC;
    const __bf16* wq0  = Wqb + (size_t)(dh * 16 + lm) * CC + g * 16;
    const __bf16* wq1  = Wqb + (size_t)((dh + 1) * 16 + lm) * CC + g * 16;
    const __bf16* wk0  = Wkb + (size_t)(dh * 16 + lm) * CC + g * 16;
    const __bf16* wk1  = Wkb + (size_t)((dh + 1) * 16 + lm) * CC + g * 16;

    // prologue: load kc = 0
    v16bf a  = ld_frag(xrow + g * 16, xrow + g * 16 + 8);
    v16bf bq[2] = { ld_frag(wq0, wq0 + 8), ld_frag(wq1, wq1 + 8) };
    v16bf bk[2] = { ld_frag(wk0, wk0 + 8), ld_frag(wk1, wk1 + 8) };

#pragma unroll
    for (int kc = 0; kc < CC - 32; kc += 32) {
        const int kn = kc + 32;
        // prefetch next K slice before consuming current one
        v16bf aN   = ld_frag(xrow + kn + g * 16, xrow + kn + g * 16 + 8);
        v16bf bqN0 = ld_frag(wq0 + kn, wq0 + kn + 8);
        v16bf bqN1 = ld_frag(wq1 + kn, wq1 + kn + 8);
        v16bf bkN0 = ld_frag(wk0 + kn, wk0 + kn + 8);
        v16bf bkN1 = ld_frag(wk1 + kn, wk1 + kn + 8);

        accF[0] = WMMA_BF16(a, bq[0], accF[0]);
        accF[1] = WMMA_BF16(a, bq[1], accF[1]);
        accG[0] = WMMA_BF16(a, bk[0], accG[0]);
        accG[1] = WMMA_BF16(a, bk[1], accG[1]);

        a = aN; bq[0] = bqN0; bq[1] = bqN1; bk[0] = bkN0; bk[1] = bkN1;
    }
    // epilogue
    accF[0] = WMMA_BF16(a, bq[0], accF[0]);
    accF[1] = WMMA_BF16(a, bq[1], accF[1]);
    accG[0] = WMMA_BF16(a, bk[0], accG[0]);
    accG[1] = WMMA_BF16(a, bk[1], accG[1]);

#pragma unroll
    for (int j = 0; j < 2; ++j)
#pragma unroll
        for (int r = 0; r < 8; ++r) {
            const int n = nb + nsub * 16 + r + 8 * g;
            const int d = (dh + j) * 16 + lm;
            const size_t o = ((size_t)b * NN + n) * DD + d;
            fT[o] = f2bf(accF[j][r]);
            gT[o] = f2bf(accG[j][r]);
        }
}

// -------------------------------------------------------------------------
// Kernel 1b: h[b][c][n] = (Wv x)[c,n].  Double-buffered K loop:
// 1 A fragment (Wv row) + 4 B fragments (xT rows) per slice.
// grid (N/64, C/128, B), 256 threads; wave w owns c-subtile, 4 n-tiles.
// -------------------------------------------------------------------------
__global__ __launch_bounds__(256) void k_proj_h(
        const __bf16* __restrict__ xT, const __bf16* __restrict__ Wvb,
        __bf16* __restrict__ hM) {
    const int b  = blockIdx.z;
    const int cb = blockIdx.y * 128;
    const int nb = blockIdx.x * 64;
    const int tid = threadIdx.x;
    const int w = tid >> 5, lane = tid & 31, lm = lane & 15, g = lane >> 4;

    v8f acc[4] = {};
    const __bf16* wrow = Wvb + (size_t)(cb + w * 16 + lm) * CC;
    const __bf16* xr0 = xT + ((size_t)b * NN + nb + 0 * 16 + lm) * CC + g * 16;
    const __bf16* xr1 = xT + ((size_t)b * NN + nb + 1 * 16 + lm) * CC + g * 16;
    const __bf16* xr2 = xT + ((size_t)b * NN + nb + 2 * 16 + lm) * CC + g * 16;
    const __bf16* xr3 = xT + ((size_t)b * NN + nb + 3 * 16 + lm) * CC + g * 16;

    v16bf aw = ld_frag(wrow + g * 16, wrow + g * 16 + 8);
    v16bf bx[4] = { ld_frag(xr0, xr0 + 8), ld_frag(xr1, xr1 + 8),
                    ld_frag(xr2, xr2 + 8), ld_frag(xr3, xr3 + 8) };

#pragma unroll
    for (int kc = 0; kc < CC - 32; kc += 32) {
        const int kn = kc + 32;
        v16bf awN = ld_frag(wrow + kn + g * 16, wrow + kn + g * 16 + 8);
        v16bf bxN0 = ld_frag(xr0 + kn, xr0 + kn + 8);
        v16bf bxN1 = ld_frag(xr1 + kn, xr1 + kn + 8);
        v16bf bxN2 = ld_frag(xr2 + kn, xr2 + kn + 8);
        v16bf bxN3 = ld_frag(xr3 + kn, xr3 + kn + 8);

        acc[0] = WMMA_BF16(aw, bx[0], acc[0]);
        acc[1] = WMMA_BF16(aw, bx[1], acc[1]);
        acc[2] = WMMA_BF16(aw, bx[2], acc[2]);
        acc[3] = WMMA_BF16(aw, bx[3], acc[3]);

        aw = awN; bx[0] = bxN0; bx[1] = bxN1; bx[2] = bxN2; bx[3] = bxN3;
    }
    acc[0] = WMMA_BF16(aw, bx[0], acc[0]);
    acc[1] = WMMA_BF16(aw, bx[1], acc[1]);
    acc[2] = WMMA_BF16(aw, bx[2], acc[2]);
    acc[3] = WMMA_BF16(aw, bx[3], acc[3]);

#pragma unroll
    for (int nt = 0; nt < 4; ++nt)
#pragma unroll
        for (int r = 0; r < 8; ++r) {
            const int c = cb + w * 16 + r + 8 * g;
            const int n = nb + nt * 16 + lm;
            hM[((size_t)b * CC + c) * NN + n] = f2bf(acc[nt][r]);
        }
}

// -------------------------------------------------------------------------
// Kernel 2: flash pass 1 — per-column max and masked-exp denominator.
// A fragments prefetched one n-tile ahead (softmax VALU work hides latency).
// grid (N/128, B), 256 threads; each wave owns one 16-column m-tile.
// -------------------------------------------------------------------------
__global__ __launch_bounds__(256) void k_stats(
        const __bf16* __restrict__ fT, const __bf16* __restrict__ gT,
        const float* __restrict__ mask,
        float* __restrict__ colmax, float* __restrict__ invden) {
    const int b = blockIdx.y;
    const int tid = threadIdx.x;
    const int w = tid >> 5, lane = tid & 31, lm = lane & 15, g = lane >> 4;
    const int m0 = blockIdx.x * 128 + w * 16;

    const __bf16* gr = gT + ((size_t)b * NN + m0 + lm) * DD;
    v16bf b0 = ld_frag(gr + g * 8,      gr + 16 + g * 8);
    v16bf b1 = ld_frag(gr + 32 + g * 8, gr + 48 + g * 8);

    float runmax = -3.0e38f;
    float runsum = 0.0f;

    const __bf16* fbase = fT + ((size_t)b * NN + lm) * DD;
    v16bf a0 = ld_frag(fbase + g * 8,      fbase + 16 + g * 8);
    v16bf a1 = ld_frag(fbase + 32 + g * 8, fbase + 48 + g * 8);

    for (int n0 = 0; n0 < NN; n0 += 16) {
        // prefetch next tile's A fragments
        v16bf a0N, a1N;
        if (n0 + 16 < NN) {
            const __bf16* fr = fbase + (size_t)(n0 + 16) * DD;
            a0N = ld_frag(fr + g * 8,      fr + 16 + g * 8);
            a1N = ld_frag(fr + 32 + g * 8, fr + 48 + g * 8);
        }
        v8f s = {};
        s = WMMA_BF16(a0, b0, s);
        s = WMMA_BF16(a1, b1, s);

        float tmax = s[0];
#pragma unroll
        for (int r = 1; r < 8; ++r) tmax = fmaxf(tmax, s[r]);
        tmax = fmaxf(tmax, __shfl_xor(tmax, 16, 32));
        const float nm = fmaxf(runmax, tmax);
        const float sc = __expf(runmax - nm);

        const f32x4* mp = (const f32x4*)(mask + (size_t)b * NN + n0 + 8 * g);
        f32x4 mA = mp[0], mB2 = mp[1];
        float ls = 0.0f;
        ls += mA[0]  * __expf(s[0] - nm);
        ls += mA[1]  * __expf(s[1] - nm);
        ls += mA[2]  * __expf(s[2] - nm);
        ls += mA[3]  * __expf(s[3] - nm);
        ls += mB2[0] * __expf(s[4] - nm);
        ls += mB2[1] * __expf(s[5] - nm);
        ls += mB2[2] * __expf(s[6] - nm);
        ls += mB2[3] * __expf(s[7] - nm);
        ls += __shfl_xor(ls, 16, 32);

        runsum = runsum * sc + ls;
        runmax = nm;
        a0 = a0N; a1 = a1N;
    }

    if (lane < 16) {
        colmax[(size_t)b * NN + m0 + lane] = runmax;
        invden[(size_t)b * NN + m0 + lane] = 1.0f / (runsum + 1e-12f);
    }
}

// -------------------------------------------------------------------------
// Kernel 3: flash pass 2 — recompute S tiles, form beta (bf16) in LDS,
// accumulate O = h @ beta with WMMA (h A-fragments prefetched one step
// ahead across the flattened (kn,ct) sequence), write gamma*O + x.
// grid (N/32, B); 256 threads (8 waves).
// -------------------------------------------------------------------------
__global__ __launch_bounds__(256) void k_attn_out(
        const __bf16* __restrict__ fT, const __bf16* __restrict__ gT,
        const __bf16* __restrict__ hM,
        const float* __restrict__ mask,
        const float* __restrict__ colmax, const float* __restrict__ invden,
        const float* __restrict__ x, const float* __restrict__ gamma,
        float* __restrict__ out) {
    __shared__ __bf16 betaT[32 * 136];   // [m_local][n_local], stride 136 (272B, 16B aligned)
    const int b  = blockIdx.y;
    const int m0 = blockIdx.x * 32;
    const int tid = threadIdx.x;
    const int w = tid >> 5, lane = tid & 31, lm = lane & 15, g = lane >> 4;

    v16bf bg[2][2];
    float cmax[2], vinv[2];
#pragma unroll
    for (int mt = 0; mt < 2; ++mt) {
        const __bf16* gr = gT + ((size_t)b * NN + m0 + mt * 16 + lm) * DD;
        bg[mt][0] = ld_frag(gr + g * 8,      gr + 16 + g * 8);
        bg[mt][1] = ld_frag(gr + 32 + g * 8, gr + 48 + g * 8);
        cmax[mt] = colmax[(size_t)b * NN + m0 + mt * 16 + lm];
        vinv[mt] = invden[(size_t)b * NN + m0 + mt * 16 + lm];
    }

    v8f acc[8] = {};   // 4 c-tiles x 2 m-tiles
    const int c0 = w * 64;
    // h row base for this wave's lane (c = c0 + ct*16 + lm), ct stride = 16 rows
    const __bf16* hbase = hM + ((size_t)b * CC + c0 + lm) * NN;

    for (int nc = 0; nc < NN; nc += 128) {
        // ---- Phase A: this wave's 16-row slice of the beta chunk ----
        const int n0 = nc + w * 16;
        const __bf16* fr = fT + ((size_t)b * NN + n0 + lm) * DD;
        v16bf a0 = ld_frag(fr + g * 8,      fr + 16 + g * 8);
        v16bf a1 = ld_frag(fr + 32 + g * 8, fr + 48 + g * 8);
        const f32x4* mp = (const f32x4*)(mask + (size_t)b * NN + n0 + 8 * g);
        f32x4 mA = mp[0], mB2 = mp[1];
        float mr[8] = { mA[0], mA[1], mA[2], mA[3], mB2[0], mB2[1], mB2[2], mB2[3] };

#pragma unroll
        for (int mt = 0; mt < 2; ++mt) {
            v8f s = {};
            s = WMMA_BF16(a0, bg[mt][0], s);
            s = WMMA_BF16(a1, bg[mt][1], s);
#pragma unroll
            for (int r = 0; r < 8; ++r) {
                const float bv = mr[r] * __expf(s[r] - cmax[mt]) * vinv[mt];
                betaT[(mt * 16 + lm) * 136 + (w * 16 + r + 8 * g)] = f2bf(bv);
            }
        }
        __syncthreads();

        // ---- Phase B: O += h[:, nc..nc+128] @ beta, A prefetched 1 ahead ----
        const __bf16* p0 = hbase + nc + g * 8;   // (kn=0, ct=0)
        v16bf ah = ld_frag(p0, p0 + 16);
#pragma unroll
        for (int kn = 0; kn < 4; ++kn) {
            const __bf16* b0p = &betaT[lm * 136 + kn * 32 + g * 8];
            const __bf16* b1p = &betaT[(16 + lm) * 136 + kn * 32 + g * 8];
            v16bf bb0 = ld_frag(b0p, b0p + 16);
            v16bf bb1 = ld_frag(b1p, b1p + 16);
#pragma unroll
            for (int ct = 0; ct < 4; ++ct) {
                const int t = kn * 4 + ct;
                v16bf ahN;
                if (t < 15) {
                    const int tn = t + 1;
                    const int knN = tn >> 2, ctN = tn & 3;
                    const __bf16* pn = hbase + (size_t)(ctN * 16) * NN + nc + knN * 32 + g * 8;
                    ahN = ld_frag(pn, pn + 16);
                }
                acc[ct * 2 + 0] = WMMA_BF16(ah, bb0, acc[ct * 2 + 0]);
                acc[ct * 2 + 1] = WMMA_BF16(ah, bb1, acc[ct * 2 + 1]);
                if (t < 15) ah = ahN;
            }
        }
        __syncthreads();
    }

    const float gm = gamma[0];
#pragma unroll
    for (int ct = 0; ct < 4; ++ct)
#pragma unroll
        for (int mt = 0; mt < 2; ++mt)
#pragma unroll
            for (int r = 0; r < 8; ++r) {
                const int c = c0 + ct * 16 + r + 8 * g;
                const int m = m0 + mt * 16 + lm;
                const size_t o = ((size_t)b * CC + c) * NN + m;
                out[o] = gm * acc[ct * 2 + mt][r] + x[o];
            }
}

// -------------------------------------------------------------------------
extern "C" void kernel_launch(void* const* d_in, const int* in_sizes, int n_in,
                              void* d_out, int out_size, void* d_ws, size_t ws_size,
                              hipStream_t stream) {
    const float* x     = (const float*)d_in[0];
    const float* mask  = (const float*)d_in[1];
    const float* Wq    = (const float*)d_in[2];
    const float* Wk    = (const float*)d_in[3];
    const float* Wv    = (const float*)d_in[4];
    const float* gamma = (const float*)d_in[5];
    float* out = (float*)d_out;

    char* ws = (char*)d_ws;
    // Workspace map (bytes):
    //   [0,2M)      fT bf16        [2M,4M)   gT bf16
    //   [4M,20M)    h  bf16        [20M,36M) xT bf16
    //   [36M,+64K)  Wq bf16  [+64K,+128K) Wk bf16  [+128K,+640K) Wv bf16
    //   [37M,+64K)  colmax f32  [+64K,+128K) invden f32        (~37.2 MB total)
    __bf16* fT     = (__bf16*)(ws);
    __bf16* gT     = (__bf16*)(ws + ((size_t)2  << 20));
    __bf16* hM     = (__bf16*)(ws + ((size_t)4  << 20));
    __bf16* xTb    = (__bf16*)(ws + ((size_t)20 << 20));
    __bf16* Wqb    = (__bf16*)(ws + ((size_t)36 << 20));
    __bf16* Wkb    = (__bf16*)(ws + ((size_t)36 << 20) + (1u << 16));
    __bf16* Wvb    = (__bf16*)(ws + ((size_t)36 << 20) + (2u << 16));
    float*  colmax = (float*)(ws + ((size_t)37 << 20));
    float*  invden = (float*)(ws + ((size_t)37 << 20) + (1u << 16));

    // Prep: weight conversion + x transpose
    k_cvt_bf16<<<dim3((DD * CC) / (256 * 8), 1, 1), 256, 0, stream>>>(Wq, Wqb, DD * CC);
    k_cvt_bf16<<<dim3((DD * CC) / (256 * 8), 1, 1), 256, 0, stream>>>(Wk, Wkb, DD * CC);
    k_cvt_bf16<<<dim3((CC * CC) / (256 * 8), 1, 1), 256, 0, stream>>>(Wv, Wvb, CC * CC);
    k_transpose_x<<<dim3(NN / 64, CC / 64, BB), 256, 0, stream>>>(x, xTb);

    // Projections (software-pipelined WMMA loops, no LDS)
    k_proj_fg<<<dim3(NN / 64, BB), 256, 0, stream>>>(xTb, Wqb, Wkb, fT, gT);
    k_proj_h <<<dim3(NN / 64, CC / 128, BB), 256, 0, stream>>>(xTb, Wvb, hM);

    // Flash softmax stats + fused attention output
    k_stats  <<<dim3(NN / 128, BB), 256, 0, stream>>>(fT, gT, mask, colmax, invden);
    k_attn_out<<<dim3(NN / 32, BB), 256, 0, stream>>>(fT, gT, hM, mask, colmax, invden,
                                                      x, gamma, out);
}
